// LinearMolaLayer_46840913330228
// MI455X (gfx1250) — matrix operand order repaired and checked by reading, not compiled, not run on hardware
//
#include <hip/hip_runtime.h>
#include <stdint.h>

#define D_MODEL 1024
#define D_OUT   1024
#define RANK    16
#define NUM_TRUE 8
#define E_TOT   10
#define SCALING 2.0f
#define T_TOTAL 16384
#define K_AUG   1152   // 1024 base + 8*16 lora rank dims

typedef __attribute__((ext_vector_type(16))) __bf16 v16bf;
typedef __attribute__((ext_vector_type(8)))  float  v8f;
typedef __attribute__((ext_vector_type(4)))  unsigned int u32x4;
typedef __attribute__((ext_vector_type(4)))  float f32x4;

union Frag16 { v16bf v; u32x4 q[2]; };

__device__ __forceinline__ unsigned short f2bf(float f) {
  unsigned int u = __float_as_uint(f);
  u += 0x7FFFu + ((u >> 16) & 1u);          // round-to-nearest-even
  return (unsigned short)(u >> 16);
}
__device__ __forceinline__ float bf2f(unsigned short s) {
  return __uint_as_float(((unsigned int)s) << 16);
}

// CDNA5 async global->LDS copy (16B per lane), tracked by ASYNCcnt.
__device__ __forceinline__ void async_load_b128(unsigned lds_off, const unsigned short* g) {
  asm volatile("global_load_async_to_lds_b128 %0, %1, off"
               :: "v"(lds_off), "v"((unsigned long long)(uintptr_t)g)
               : "memory");
}

// ---------------- Kernel 1a: Wall = [base_w | lora_B] -> bf16 ----------------
__global__ void prep_weights(const float* __restrict__ base_w,
                             const float* __restrict__ lora_B,
                             unsigned short* __restrict__ Wbf) {
  int idx = blockIdx.x * blockDim.x + threadIdx.x;
  if (idx >= D_OUT * K_AUG) return;
  int n = idx / K_AUG, k = idx % K_AUG;
  float v;
  if (k < D_MODEL) {
    v = base_w[(size_t)n * D_MODEL + k];
  } else {
    int q = k - D_MODEL;
    int e = q >> 4, r = q & 15;
    v = lora_B[((size_t)e * D_OUT + n) * RANK + r];
  }
  Wbf[idx] = f2bf(v);
}

// ---------------- Kernel 1b: lora_A -> bf16 (flat [128][1024]) ---------------
__global__ void prep_a(const float* __restrict__ lora_A,
                       unsigned short* __restrict__ Abf) {
  int idx = blockIdx.x * blockDim.x + threadIdx.x;
  if (idx >= NUM_TRUE * RANK * D_MODEL) return;
  Abf[idx] = f2bf(lora_A[idx]);
}

// ------------- Kernel 2: router + z (WMMA), emits Xall[T,1152] bf16 ----------
#define XS_S 1032   // padded LDS row stride (elements) to avoid bank conflicts
__global__ __launch_bounds__(128)
void router_z(const float* __restrict__ x,
              const float* __restrict__ gate_w,
              const unsigned short* __restrict__ Abf,
              unsigned short* __restrict__ Xall) {
  __shared__ __align__(16) unsigned short xs[16 * XS_S];
  __shared__ float logits_s[16][E_TOT];
  __shared__ float ew_s[16][NUM_TRUE];

  const int tid  = threadIdx.x;
  const int lane = tid & 31;
  const int wave = tid >> 5;
  const int t0   = blockIdx.x * 16;

  // stage x -> bf16 LDS and write bf16 x into Xall[:, 0:1024]
  // x is read exactly once overall -> non-temporal loads keep L2 for Xall/Wbf
  for (int j = 0; j < 32; ++j) {
    int fi  = tid + j * 128;            // 4096 float4 chunks
    int row = fi >> 8;                  // 256 float4 per row
    int c4  = fi & 255;
    const f32x4 xv = __builtin_nontemporal_load(
        (const f32x4*)(x + (size_t)(t0 + row) * D_MODEL + c4 * 4));
    uint2 p;
    p.x = (unsigned)f2bf(xv.x) | ((unsigned)f2bf(xv.y) << 16);
    p.y = (unsigned)f2bf(xv.z) | ((unsigned)f2bf(xv.w) << 16);
    *(uint2*)(&xs[row * XS_S + c4 * 4]) = p;
    *(uint2*)(Xall + (size_t)(t0 + row) * K_AUG + c4 * 4) = p;
  }
  __syncthreads();

  // router logits, f32 accumulate
  for (int p = tid; p < 16 * E_TOT; p += 128) {
    int t = p / E_TOT, e = p % E_TOT;
    const float* gw = gate_w + (size_t)e * D_MODEL;
    const unsigned short* xr = &xs[t * XS_S];
    float acc = 0.f;
    for (int k = 0; k < D_MODEL; ++k) acc += bf2f(xr[k]) * gw[k];
    logits_s[t][e] = acc;
  }
  __syncthreads();

  // softmax + top-2 + null-mask + renormalize (one thread per token)
  if (tid < 16) {
    const int t = tid;
    float mx = logits_s[t][0];
    for (int e = 1; e < E_TOT; ++e) mx = fmaxf(mx, logits_s[t][e]);
    float pr[E_TOT], se = 0.f;
    for (int e = 0; e < E_TOT; ++e) { pr[e] = __expf(logits_s[t][e] - mx); se += pr[e]; }
    const float inv = 1.f / se;
    int i1 = 0; float v1 = pr[0] * inv;
    for (int e = 1; e < E_TOT; ++e) { float v = pr[e] * inv; if (v > v1) { v1 = v; i1 = e; } }
    int i2 = -1; float v2 = -1.f;
    for (int e = 0; e < E_TOT; ++e) {
      if (e == i1) continue;
      float v = pr[e] * inv; if (v > v2) { v2 = v; i2 = e; }
    }
    float w1 = (i1 < NUM_TRUE) ? v1 : 0.f;
    float w2 = (i2 < NUM_TRUE) ? v2 : 0.f;
    float den = w1 + w2;
    float s = (den > 0.f) ? (1.f / den) : 0.f;
    for (int e = 0; e < NUM_TRUE; ++e) ew_s[t][e] = 0.f;
    if (i1 < NUM_TRUE) ew_s[t][i1] += w1 * s;
    if (i2 >= 0 && i2 < NUM_TRUE) ew_s[t][i2] += w2 * s;
  }
  __syncthreads();

  // z = x @ A^T via WMMA; each wave owns 2 experts (one 16x16 N-tile == one expert)
  const int arow  = lane & 15;
  const int akoff = (lane < 16) ? 0 : 8;    // A layout: lane<16 K{0..7,16..23}
  const int bkoff = (lane < 16) ? 0 : 8;    // B layout: same as A, lane<16 K{0..7,16..23}
  for (int nt = wave * 2; nt < wave * 2 + 2; ++nt) {
    v8f acc = {};
    for (int k0 = 0; k0 < D_MODEL; k0 += 32) {
      Frag16 a, b;
      const unsigned short* ap = &xs[arow * XS_S + k0 + akoff];
      a.q[0] = *(const u32x4*)(ap);
      a.q[1] = *(const u32x4*)(ap + 16);
      const unsigned short* bp = Abf + (size_t)(nt * 16 + arow) * D_MODEL + k0 + bkoff;
      b.q[0] = *(const u32x4*)(bp);
      b.q[1] = *(const u32x4*)(bp + 16);
      acc = __builtin_amdgcn_wmma_f32_16x16x32_bf16(false, a.v, false, b.v,
                                                    (short)0, acc, false, false);
    }
    // scale by SCALING*ew[t,e] and emit augmented columns of Xall
    const int r = lane & 15;
    const int rbase = (lane < 16) ? 0 : 8;   // C/D: lanes>=16 hold rows M=8..15
    for (int i = 0; i < 8; ++i) {
      int tl = rbase + i;
      float v = acc[i] * (SCALING * ew_s[tl][nt]);
      Xall[(size_t)(t0 + tl) * K_AUG + D_MODEL + nt * 16 + r] = f2bf(v);
    }
  }
}

// -------- Kernel 3: out = Xall @ Wall^T + bias  (T=16384, N=1024, K=1152) ----
// Double-buffered LDS tiles filled by async global->LDS DMA (ASYNCcnt), so the
// loads for tile k+1 overlap the WMMAs of tile k.
#define TS 40   // padded LDS k-stride (elements)
#define NKT (K_AUG / 32)
__global__ __launch_bounds__(256)
void gemm_main(const unsigned short* __restrict__ Xall,
               const unsigned short* __restrict__ Wbf,
               const float* __restrict__ base_b,
               float* __restrict__ out) {
  __shared__ __align__(16) unsigned short Xs[2][128 * TS];
  __shared__ __align__(16) unsigned short Ws[2][128 * TS];
  const int tid  = threadIdx.x;
  const int lane = tid & 31;
  const int wave = tid >> 5;
  const int wy = wave >> 1;             // 4 wave-rows  (M)
  const int wx = wave & 1;              // 2 wave-cols  (N)
  const int m0 = blockIdx.y * 128;
  const int n0 = blockIdx.x * 128;

  // per-thread tile-staging addresses: 512 x 16B chunks per matrix, 2 per thread
  const int c0 = tid, c1 = tid + 256;
  const int r0 = c0 >> 2, cx0 = (c0 & 3) * 8;
  const int r1 = c1 >> 2, cx1 = (c1 & 3) * 8;
  const unsigned short* Xg0 = Xall + (size_t)(m0 + r0) * K_AUG + cx0;
  const unsigned short* Xg1 = Xall + (size_t)(m0 + r1) * K_AUG + cx1;
  const unsigned short* Wg0 = Wbf  + (size_t)(n0 + r0) * K_AUG + cx0;
  const unsigned short* Wg1 = Wbf  + (size_t)(n0 + r1) * K_AUG + cx1;
  unsigned xo0[2], xo1[2], wo0[2], wo1[2];
  for (int b = 0; b < 2; ++b) {
    xo0[b] = (unsigned)(uintptr_t)&Xs[b][r0 * TS + cx0];
    xo1[b] = (unsigned)(uintptr_t)&Xs[b][r1 * TS + cx1];
    wo0[b] = (unsigned)(uintptr_t)&Ws[b][r0 * TS + cx0];
    wo1[b] = (unsigned)(uintptr_t)&Ws[b][r1 * TS + cx1];
  }

  v8f zero = {};
  v8f acc[2][4];
  for (int mt = 0; mt < 2; ++mt)
    for (int nt = 0; nt < 4; ++nt) acc[mt][nt] = zero;

  const int arow  = lane & 15;
  const int akoff = (lane < 16) ? 0 : 8;
  const int bkoff = (lane < 16) ? 0 : 8;

  // prologue: async-fill buffer 0 with k-tile 0 (4 async instructions / wave)
  async_load_b128(xo0[0], Xg0);
  async_load_b128(xo1[0], Xg1);
  async_load_b128(wo0[0], Wg0);
  async_load_b128(wo1[0], Wg1);

  for (int kt = 0; kt < NKT; ++kt) {
    const int cur = kt & 1;
    if (kt + 1 < NKT) {
      const int k1 = (kt + 1) * 32;
      async_load_b128(xo0[cur ^ 1], Xg0 + k1);
      async_load_b128(xo1[cur ^ 1], Xg1 + k1);
      async_load_b128(wo0[cur ^ 1], Wg0 + k1);
      async_load_b128(wo1[cur ^ 1], Wg1 + k1);
      // 8 outstanding; retire the 4 belonging to the current tile (in-order)
      asm volatile("s_wait_asynccnt 0x4" ::: "memory");
    } else {
      asm volatile("s_wait_asynccnt 0x0" ::: "memory");
    }
    __syncthreads();

    Frag16 a[2], bfr[4];
    for (int mt = 0; mt < 2; ++mt) {
      const unsigned short* p = &Xs[cur][(wy * 32 + mt * 16 + arow) * TS + akoff];
      a[mt].q[0] = *(const u32x4*)(p);
      a[mt].q[1] = *(const u32x4*)(p + 16);
    }
    for (int nt = 0; nt < 4; ++nt) {
      const unsigned short* p = &Ws[cur][(wx * 64 + nt * 16 + arow) * TS + bkoff];
      bfr[nt].q[0] = *(const u32x4*)(p);
      bfr[nt].q[1] = *(const u32x4*)(p + 16);
    }
    for (int mt = 0; mt < 2; ++mt)
      for (int nt = 0; nt < 4; ++nt)
        acc[mt][nt] = __builtin_amdgcn_wmma_f32_16x16x32_bf16(
            false, a[mt].v, false, bfr[nt].v, (short)0, acc[mt][nt], false, false);

    __syncthreads();   // all waves done reading 'cur' before it is refilled
  }

  // epilogue: add bias, non-temporal f32 stores (out is write-once)
  const int rbase = (lane < 16) ? 0 : 8;
  for (int nt = 0; nt < 4; ++nt) {
    int n = n0 + wx * 64 + nt * 16 + (lane & 15);
    float bias = base_b[n];
    for (int mt = 0; mt < 2; ++mt) {
      int mbase = m0 + wy * 32 + mt * 16 + rbase;
      v8f c = acc[mt][nt];
      for (int i = 0; i < 8; ++i)
        __builtin_nontemporal_store(c[i] + bias, &out[(size_t)(mbase + i) * D_OUT + n]);
    }
  }
}

extern "C" void kernel_launch(void* const* d_in, const int* in_sizes, int n_in,
                              void* d_out, int out_size, void* d_ws, size_t ws_size,
                              hipStream_t stream) {
  const float* x      = (const float*)d_in[0];
  const float* base_w = (const float*)d_in[1];
  const float* base_b = (const float*)d_in[2];
  const float* gate_w = (const float*)d_in[3];
  const float* lora_A = (const float*)d_in[4];
  const float* lora_B = (const float*)d_in[5];
  float* out = (float*)d_out;

  // workspace layout (all 16B aligned): Wbf | Abf | Xall  (~40.4 MB total)
  unsigned short* Wbf  = (unsigned short*)d_ws;                        // 1024*1152
  unsigned short* Abf  = Wbf + (size_t)D_OUT * K_AUG;                  // 128*1024
  unsigned short* Xall = Abf + (size_t)NUM_TRUE * RANK * D_MODEL;      // 16384*1152

  prep_weights<<<(D_OUT * K_AUG + 255) / 256, 256, 0, stream>>>(base_w, lora_B, Wbf);
  prep_a<<<(NUM_TRUE * RANK * D_MODEL + 255) / 256, 256, 0, stream>>>(lora_A, Abf);
  router_z<<<T_TOTAL / 16, 128, 0, stream>>>(x, gate_w, Abf, Xall);
  gemm_main<<<dim3(D_OUT / 128, T_TOTAL / 128), 256, 0, stream>>>(Xall, Wbf, base_b, out);
}
